// GraphAggregator_leaky_9320079033257
// MI455X (gfx1250) — matrix operand order repaired and checked by reading, not compiled, not run on hardware
//
#include <hip/hip_runtime.h>

// ---------------------------------------------------------------------------
// Problem constants (match reference)
// ---------------------------------------------------------------------------
#define N_NODES   200000
#define D_IN      256
#define H1        512
#define D_G       256
#define D_G2      512
#define T1        256
#define T2        128
#define N_GRAPHS  2000

// WMMA vector types (CDNA5, wave32)
typedef __attribute__((ext_vector_type(16))) __bf16 v16bf;
typedef __attribute__((ext_vector_type(8)))  float  v8f;

union Frag16 {
    v16bf bf;
    uint4 q[2];   // two 16-byte chunks = 16 bf16 values
};

__device__ __forceinline__ unsigned short f32_to_bf16(float f) {
    unsigned int u = __float_as_uint(f);
    unsigned int r = u + 0x7FFFu + ((u >> 16) & 1u);   // round-to-nearest-even
    return (unsigned short)(r >> 16);
}

__device__ __forceinline__ unsigned int pack2_bf16(float a, float b) {
    return (unsigned int)f32_to_bf16(a) | ((unsigned int)f32_to_bf16(b) << 16);
}

__device__ __forceinline__ v8f wmma_bf16(const Frag16& a, const Frag16& b, v8f c) {
    // (neg_a, A, neg_b, B, c_mod, C, reuse_a, reuse_b)
    return __builtin_amdgcn_wmma_f32_16x16x32_bf16(false, a.bf, false, b.bf,
                                                   (short)0, c, false, false);
}

// A-fragment (16x32 bf16) from row-major bf16 buffer with row stride `strideK`.
// ISA layout: lanes 0-15 row M=lane, elems = K{k0..k0+7, k0+16..k0+23};
//             lanes 16-31 row M=lane-16, elems = K{k0+8..15, k0+24..31}.
__device__ __forceinline__ Frag16 load_a(const unsigned short* base, int row,
                                         int strideK, int k0, int lrow, int lh) {
    Frag16 f;
    const unsigned short* p = base + (size_t)(row + lrow) * strideK + k0 + lh * 8;
    f.q[0] = *(const uint4*)(p);
    f.q[1] = *(const uint4*)(p + 16);
    return f;
}

// B-fragment (32x16 bf16) from pre-transposed weights Wt[N][K] (row-major, K contig).
// ISA layout: lane = column N; lanes 0-15 hold K{0..7, 16..23}, lanes 16-31 hold K{8..15, 24..31}.
__device__ __forceinline__ Frag16 load_b(const unsigned short* wt, int n0, int K,
                                         int k0, int lrow, int lh) {
    Frag16 f;
    const unsigned short* p = wt + (size_t)(n0 + lrow) * K + k0 + lh * 8;
    f.q[0] = *(const uint4*)(p);
    f.q[1] = *(const uint4*)(p + 16);
    return f;
}

// ---------------------------------------------------------------------------
// Prep kernels
// ---------------------------------------------------------------------------
__global__ void zero_f32_kernel(float* __restrict__ p, int n4) {
    int i = blockIdx.x * 256 + threadIdx.x;
    if (i < n4) *(float4*)(p + i * 4) = make_float4(0.f, 0.f, 0.f, 0.f);
}

// in: f32 [K][N] row-major  ->  out: bf16 [N][K] row-major
__global__ void transpose_bf16_kernel(const float* __restrict__ in,
                                      unsigned short* __restrict__ out,
                                      int K, int N) {
    int idx = blockIdx.x * 256 + threadIdx.x;
    if (idx < K * N) {
        int k = idx / N;
        int n = idx - k * N;
        out[(size_t)n * K + k] = f32_to_bf16(in[idx]);
    }
}

// ---------------------------------------------------------------------------
// Kernel 1: fused  h = relu(X@W1+b1); g = h@W2+b2; u = g[:,256:]*sigmoid(g[:,:256])
//           atomic segment-sum of u into gstates[graph_idx]
// 64 nodes per block, 256 threads (8 waves); B fragments reused across 4 M-tiles
// Dynamic LDS: sA 32KB | sH 64KB | sIdx 256B  (96.25 KB of the 320 KB WGP LDS)
// ---------------------------------------------------------------------------
#define NODE_LDS_BYTES (64 * D_IN * 2 + 64 * H1 * 2 + 64 * 4)

__global__ void __launch_bounds__(256)
node_mlp_kernel(const float* __restrict__ X,
                const int* __restrict__ gidx,
                const unsigned short* __restrict__ W1t,   // [H1][D_IN] bf16
                const float* __restrict__ b1,
                const unsigned short* __restrict__ W2t,   // [D_G2][H1] bf16
                const float* __restrict__ b2,
                float* __restrict__ gstates)              // [N_GRAPHS][D_G]
{
    extern __shared__ char smem[];
    unsigned short* sA   = (unsigned short*)smem;                       // 64 x 256 bf16
    unsigned short* sH   = (unsigned short*)(smem + 64 * D_IN * 2);     // 64 x 512 bf16
    int*            sIdx = (int*)(smem + 64 * D_IN * 2 + 64 * H1 * 2);  // 64 ints

    const int tid  = threadIdx.x;
    const int wave = tid >> 5;
    const int lane = tid & 31;
    const int lh   = lane >> 4;     // half-wave select
    const int lrow = lane & 15;
    const size_t nodeBase = (size_t)blockIdx.x * 64;

    // ---- stage X tile (f32 -> bf16, packed 8-byte LDS stores) ----
    {
        const float4* src = (const float4*)(X + nodeBase * D_IN);
        for (int i = tid; i < 64 * D_IN / 4; i += 256) {
            float4 v = src[i];
            *(uint2*)(sA + i * 4) = make_uint2(pack2_bf16(v.x, v.y),
                                              pack2_bf16(v.z, v.w));
        }
        if (tid < 64) sIdx[tid] = gidx[nodeBase + tid];
    }
    __syncthreads();

    // ---- GEMM1: [64 x 256] @ [256 x 512] -> relu -> sH ----
    // Each wave owns 4 column tiles; B loaded once per k-step, reused by 4 M-tiles.
#pragma unroll 1
    for (int j = 0; j < 4; ++j) {
        const int nt = wave + 8 * j;               // 0..31
        v8f acc[4] = {{}, {}, {}, {}};
        for (int k0 = 0; k0 < D_IN; k0 += 32) {
            Frag16 b = load_b(W1t, nt * 16, D_IN, k0, lrow, lh);
            if (k0 + 32 < D_IN)
                __builtin_prefetch(W1t + (size_t)(nt * 16 + lrow) * D_IN
                                       + (k0 + 32) + lh * 16, 0, 0);
#pragma unroll
            for (int mt = 0; mt < 4; ++mt) {
                Frag16 a = load_a(sA, mt * 16, D_IN, k0, lrow, lh);
                acc[mt] = wmma_bf16(a, b, acc[mt]);
            }
        }
        const int   col  = nt * 16 + lrow;
        const float bias = b1[col];
#pragma unroll
        for (int mt = 0; mt < 4; ++mt)
#pragma unroll
            for (int r = 0; r < 8; ++r) {
                float v = acc[mt][r] + bias;
                v = v > 0.f ? v : 0.f;
                sH[(mt * 16 + lh * 8 + r) * H1 + col] = f32_to_bf16(v);
            }
    }
    __syncthreads();

    // ---- GEMM2 (gate/value pair) + sigmoid gate + atomic segment sum ----
    // Each wave owns 2 gate-column tiles; B pair loaded once per k-step -> 8 WMMAs.
#pragma unroll 1
    for (int j = 0; j < 2; ++j) {
        const int ng = wave * 2 + j;               // 0..15 -> gate cols N = ng*16..
        v8f accG[4] = {{}, {}, {}, {}};
        v8f accV[4] = {{}, {}, {}, {}};
        for (int k0 = 0; k0 < H1; k0 += 32) {
            Frag16 bg = load_b(W2t, ng * 16, H1, k0, lrow, lh);
            Frag16 bv = load_b(W2t, D_G + ng * 16, H1, k0, lrow, lh);
            if (k0 + 32 < H1) {
                __builtin_prefetch(W2t + (size_t)(ng * 16 + lrow) * H1
                                       + (k0 + 32) + lh * 16, 0, 0);
                __builtin_prefetch(W2t + (size_t)(D_G + ng * 16 + lrow) * H1
                                       + (k0 + 32) + lh * 16, 0, 0);
            }
#pragma unroll
            for (int mt = 0; mt < 4; ++mt) {
                Frag16 a = load_a(sH, mt * 16, H1, k0, lrow, lh);
                accG[mt] = wmma_bf16(a, bg, accG[mt]);
                accV[mt] = wmma_bf16(a, bv, accV[mt]);
            }
        }
        const int   col = ng * 16 + lrow;          // 0..255
        const float bg_ = b2[col];
        const float bv_ = b2[col + D_G];
#pragma unroll
        for (int mt = 0; mt < 4; ++mt)
#pragma unroll
            for (int r = 0; r < 8; ++r) {
                const int m = mt * 16 + lh * 8 + r;
                float gval = accG[mt][r] + bg_;
                float vval = accV[mt][r] + bv_;
                float u = vval * (1.f / (1.f + __expf(-gval)));
                atomicAdd(&gstates[(size_t)sIdx[m] * D_G + col], u);
            }
    }
}

// ---------------------------------------------------------------------------
// Kernel 2: graph MLP2:  e = leaky(G@gW1+gb1)  (out1);  o = e@gW2+gb2 (out2)
// 32 graphs per block, 256 threads (8 waves)
// ---------------------------------------------------------------------------
__global__ void __launch_bounds__(256)
graph_mlp_kernel(const float* __restrict__ G,                 // [N_GRAPHS][D_G]
                 const unsigned short* __restrict__ gW1t,     // [T1][D_G] bf16
                 const float* __restrict__ gb1,
                 const unsigned short* __restrict__ gW2t,     // [T2][T1] bf16
                 const float* __restrict__ gb2,
                 float* __restrict__ out1,                    // [N_GRAPHS][T1]
                 float* __restrict__ out2)                    // [N_GRAPHS][T2]
{
    __shared__ unsigned short sG[32 * D_G];   // 16 KB
    __shared__ unsigned short sE[32 * T1];    // 16 KB

    const int tid  = threadIdx.x;
    const int wave = tid >> 5;
    const int lane = tid & 31;
    const int lh   = lane >> 4;
    const int lrow = lane & 15;
    const int gBase = blockIdx.x * 32;

    // ---- stage graph-state tile (clamped rows; stores guarded later) ----
    for (int i = tid; i < 32 * D_G / 4; i += 256) {
        int row = i >> 6;               // 64 float4 per row
        int c4  = (i & 63) << 2;
        int grow = gBase + row;
        if (grow > N_GRAPHS - 1) grow = N_GRAPHS - 1;
        float4 v = *(const float4*)(G + (size_t)grow * D_G + c4);
        *(uint2*)(sG + i * 4) = make_uint2(pack2_bf16(v.x, v.y),
                                           pack2_bf16(v.z, v.w));
    }
    __syncthreads();

    // ---- GEMM3: [32 x 256] @ [256 x 256], LeakyReLU ----
#pragma unroll 1
    for (int j = 0; j < 2; ++j) {
        const int nt = wave + 8 * j;               // 0..15
        v8f acc[2] = {{}, {}};
        for (int k0 = 0; k0 < D_G; k0 += 32) {
            Frag16 b = load_b(gW1t, nt * 16, D_G, k0, lrow, lh);
#pragma unroll
            for (int mt = 0; mt < 2; ++mt) {
                Frag16 a = load_a(sG, mt * 16, D_G, k0, lrow, lh);
                acc[mt] = wmma_bf16(a, b, acc[mt]);
            }
        }
        const int   col  = nt * 16 + lrow;
        const float bias = gb1[col];
#pragma unroll
        for (int mt = 0; mt < 2; ++mt)
#pragma unroll
            for (int r = 0; r < 8; ++r) {
                const int m = mt * 16 + lh * 8 + r;
                float v = acc[mt][r] + bias;
                v = v > 0.f ? v : 0.01f * v;
                const int grow = gBase + m;
                if (grow < N_GRAPHS) out1[(size_t)grow * T1 + col] = v;
                sE[m * T1 + col] = f32_to_bf16(v);
            }
    }
    __syncthreads();

    // ---- GEMM4: [32 x 256] @ [256 x 128] ----
    {
        const int nt = wave;                       // 0..7
        v8f acc[2] = {{}, {}};
        for (int k0 = 0; k0 < T1; k0 += 32) {
            Frag16 b = load_b(gW2t, nt * 16, T1, k0, lrow, lh);
#pragma unroll
            for (int mt = 0; mt < 2; ++mt) {
                Frag16 a = load_a(sE, mt * 16, T1, k0, lrow, lh);
                acc[mt] = wmma_bf16(a, b, acc[mt]);
            }
        }
        const int   col  = nt * 16 + lrow;
        const float bias = gb2[col];
#pragma unroll
        for (int mt = 0; mt < 2; ++mt)
#pragma unroll
            for (int r = 0; r < 8; ++r) {
                const int m = mt * 16 + lh * 8 + r;
                const int grow = gBase + m;
                if (grow < N_GRAPHS) out2[(size_t)grow * T2 + col] = acc[mt][r] + bias;
            }
    }
}

// ---------------------------------------------------------------------------
// Launch
// ---------------------------------------------------------------------------
extern "C" void kernel_launch(void* const* d_in, const int* in_sizes, int n_in,
                              void* d_out, int out_size, void* d_ws, size_t ws_size,
                              hipStream_t stream) {
    const float* X    = (const float*)d_in[0];
    const int*   gidx = (const int*)d_in[1];
    // d_in[2] = n_graphs scalar (fixed at N_GRAPHS)
    const float* W1   = (const float*)d_in[3];
    const float* b1   = (const float*)d_in[4];
    const float* W2   = (const float*)d_in[5];
    const float* b2   = (const float*)d_in[6];
    const float* gW1  = (const float*)d_in[7];
    const float* gb1  = (const float*)d_in[8];
    const float* gW2  = (const float*)d_in[9];
    const float* gb2  = (const float*)d_in[10];

    // workspace layout
    char* ws = (char*)d_ws;
    float*          gstates = (float*)ws;                                  // 2,048,000 B
    unsigned short* W1t  = (unsigned short*)(ws + 2048000);                //   262,144 B
    unsigned short* W2t  = (unsigned short*)(ws + 2048000 + 262144);       //   524,288 B
    unsigned short* gW1t = (unsigned short*)(ws + 2048000 + 262144 + 524288);          // 131,072 B
    unsigned short* gW2t = (unsigned short*)(ws + 2048000 + 262144 + 524288 + 131072); //  65,536 B

    float* out1 = (float*)d_out;
    float* out2 = out1 + (size_t)N_GRAPHS * T1;

    // 1) zero per-graph accumulator (deterministic per call)
    zero_f32_kernel<<<(N_GRAPHS * D_G / 4 + 255) / 256, 256, 0, stream>>>(
        gstates, N_GRAPHS * D_G / 4);

    // 2) weight convert/transpose to bf16 [N][K]
    transpose_bf16_kernel<<<(D_IN * H1 + 255) / 256, 256, 0, stream>>>(W1, W1t, D_IN, H1);
    transpose_bf16_kernel<<<(H1 * D_G2 + 255) / 256, 256, 0, stream>>>(W2, W2t, H1, D_G2);
    transpose_bf16_kernel<<<(D_G * T1 + 255) / 256, 256, 0, stream>>>(gW1, gW1t, D_G, T1);
    transpose_bf16_kernel<<<(T1 * T2 + 255) / 256, 256, 0, stream>>>(gW2, gW2t, T1, T2);

    // 3) fused node MLP + gating + atomic segment sum (64 nodes/block)
    node_mlp_kernel<<<N_NODES / 64, 256, NODE_LDS_BYTES, stream>>>(
        X, gidx, W1t, b1, W2t, b2, gstates);

    // 4) graph MLP -> both outputs
    graph_mlp_kernel<<<(N_GRAPHS + 31) / 32, 256, 0, stream>>>(
        gstates, gW1t, gb1, gW2t, gb2, out1, out2);
}
